// Attention_14843406975553
// MI455X (gfx1250) — matrix operand order repaired and checked by reading, not compiled, not run on hardware
//
#include <hip/hip_runtime.h>

typedef __bf16 bf16_t;
typedef __attribute__((ext_vector_type(16))) __bf16 v16bf;
typedef __attribute__((ext_vector_type(8)))  __bf16 v8bf;
typedef __attribute__((ext_vector_type(4)))  __bf16 v4bf;
typedef __attribute__((ext_vector_type(8)))  float  v8f;

// ---------- WMMA fragment helpers (contiguous k = 16*(lane/16)+e, same for A and B) ----------

// A-matrix 16x32 bf16 from row-major buffer (row stride ld elements).
// lane L: M = L%16, kbase = 16*(L/16); elems 0..7 -> K=kbase..kbase+7,
// elems 8..15 -> K=kbase+8..kbase+15  => two contiguous 16B chunks.
static __device__ __forceinline__ v16bf load_a16x32(const bf16_t* base, int ld, int lane) {
  const int m  = lane & 15;
  const int kb = (lane >> 4) << 4;
  const bf16_t* p = base + m * ld + kb;
  v8bf lo = *(const v8bf*)(p);
  v8bf hi = *(const v8bf*)(p + 8);
  v16bf r;
#pragma unroll
  for (int i = 0; i < 8; ++i) { r[i] = lo[i]; r[i + 8] = hi[i]; }
  return r;
}

// B-matrix 32x16 bf16: lane L holds column N = L%16, K = 16*(L/16) + e
// => one contiguous 32B chunk per lane when storage is N-major (K contiguous).
static __device__ __forceinline__ v16bf load_b32x16(const bf16_t* base, int ld, int lane) {
  const int n  = lane & 15;
  const int kb = (lane >> 4) << 4;
  return *(const v16bf*)(base + n * ld + kb);
}

static __device__ __forceinline__ v8f wmma_bf16(v16bf a, v16bf b, v8f c) {
  return __builtin_amdgcn_wmma_f32_16x16x32_bf16(false, a, false, b, (short)0, c,
                                                 false, false);
}

// ---------- Pre-pass: weight transpose->bf16 + bias materialization ----------

__global__ void prep_kernel(const float* __restrict__ wqkv,   // [256][768]
                            const float* __restrict__ wout,   // [256][256]
                            const float* __restrict__ btab,   // [209][8]
                            const int*   __restrict__ rpi,    // [60][60]
                            bf16_t* __restrict__ wq_t,        // [768][256]
                            bf16_t* __restrict__ wo_t,        // [256][256]
                            float*  __restrict__ bias_full) { // [8][64][64]
  const int i = blockIdx.x * blockDim.x + threadIdx.x;
  if (i < 768 * 256) {
    const int nn = i >> 8, c = i & 255;
    wq_t[i] = (bf16_t)wqkv[c * 768 + nn];
  }
  if (i < 256 * 256) {
    const int nn = i >> 8, c = i & 255;
    wo_t[i] = (bf16_t)wout[c * 256 + nn];
  }
  if (i < 8 * 64 * 64) {
    const int hh = i >> 12, row = (i >> 6) & 63, col = i & 63;
    float v = -1.0e30f;                       // padding columns kill softmax
    if (row < 60 && col < 60) v = btab[rpi[row * 60 + col] * 8 + hh];
    bias_full[i] = v;
  }
}

// ---------- Fused windowed attention: one block per window ----------

__global__ __launch_bounds__(256, 1) void attn_win_kernel(
    const float*  __restrict__ x,          // [4096][60][256]
    const bf16_t* __restrict__ wq_t,       // [768][256]
    const bf16_t* __restrict__ wo_t,       // [256][256]
    const float*  __restrict__ bias_full,  // [8][64][64]
    float*        __restrict__ out) {      // [4096][60][256]
  extern __shared__ char smem_raw[];
  bf16_t* xb = (bf16_t*)smem_raw;   // [64][256] staged X          32 KB
  bf16_t* qs = xb + 64 * 256;       // [64][256] Q (pre-scaled)    32 KB
  bf16_t* ks = qs + 64 * 256;       // [64][256] K                 32 KB
  bf16_t* vt = ks + 64 * 256;       // [256][64] V transposed      32 KB
  bf16_t* os = vt + 256 * 64;       // [64][256] attention output  32 KB
  bf16_t* ps = os + 64 * 256;       // [8][64][64] P (softmax)     64 KB

  const int tid  = threadIdx.x;
  const int lane = tid & 31;
  const int wave = tid >> 5;
  const int blk  = blockIdx.x;
  const int nl   = lane & 15;          // C-matrix column within tile
  const int r0   = (lane >> 4) << 3;   // C-matrix row base within tile
  const float scale = 0.17677669529663687f;  // 1/sqrt(32)

  // ---- stage X: 60x256 f32 -> 64x256 bf16 (vectorized), pad rows zero ----
  {
    const float4* xg4 = (const float4*)(x + (size_t)blk * (60 * 256));
    for (int i = tid; i < 64 * 256 / 4; i += 256) {
      const int t = i >> 6;                       // 64 float4 per row
      float4 v = make_float4(0.f, 0.f, 0.f, 0.f);
      if (t < 60) v = xg4[i];
      v4bf p;
      p[0] = (bf16_t)v.x; p[1] = (bf16_t)v.y; p[2] = (bf16_t)v.z; p[3] = (bf16_t)v.w;
      *(v4bf*)(xb + 4 * i) = p;                   // 8B aligned LDS store
    }
  }
  __syncthreads();

  // ---- QKV GEMM: [64x256] x [256x768]; 48 N-tiles split across 8 waves ----
  for (int nt = wave; nt < 48; nt += 8) {
    v8f acc[4] = {};
    const bf16_t* wb = wq_t + nt * 16 * 256;
#pragma unroll
    for (int k = 0; k < 8; ++k) {
      v16bf bfr = load_b32x16(wb + k * 32, 256, lane);
      v16bf af[4];
#pragma unroll
      for (int mt = 0; mt < 4; ++mt)              // batch A loads: one DS wait
        af[mt] = load_a16x32(xb + mt * 16 * 256 + k * 32, 256, lane);
#pragma unroll
      for (int mt = 0; mt < 4; ++mt)              // 4 back-to-back WMMAs
        acc[mt] = wmma_bf16(af[mt], bfr, acc[mt]);
    }
    if (nt < 16) {                       // Q: row-major, fold in softmax scale
#pragma unroll
      for (int mt = 0; mt < 4; ++mt)
#pragma unroll
        for (int j = 0; j < 8; ++j)
          qs[(mt * 16 + r0 + j) * 256 + nt * 16 + nl] = (bf16_t)(acc[mt][j] * scale);
    } else if (nt < 32) {                // K: row-major
#pragma unroll
      for (int mt = 0; mt < 4; ++mt)
#pragma unroll
        for (int j = 0; j < 8; ++j)
          ks[(mt * 16 + r0 + j) * 256 + (nt - 16) * 16 + nl] = (bf16_t)acc[mt][j];
    } else {                             // V: store transposed [feature][token]
      const int f0 = (nt - 32) * 16;
#pragma unroll
      for (int mt = 0; mt < 4; ++mt)
#pragma unroll
        for (int j = 0; j < 8; ++j)
          vt[(f0 + nl) * 64 + mt * 16 + r0 + j] = (bf16_t)acc[mt][j];
    }
  }
  __syncthreads();

  // ---- attention: wave w owns head w ----
  {
    const int h = wave;
    const bf16_t* qh = qs + h * 32;
    const bf16_t* kh = ks + h * 32;
    bf16_t* ph = ps + h * 64 * 64;
    const float* bh = bias_full + h * 64 * 64;

    // K^T B-fragments: S[i,j] = sum_d q[i,d] k[j,d]; B[d,j] = k[j,d] row-major.
    v16bf bk[4];
#pragma unroll
    for (int ntv = 0; ntv < 4; ++ntv)
      bk[ntv] = load_b32x16(kh + ntv * 16 * 256, 256, lane);

#pragma unroll
    for (int mt = 0; mt < 4; ++mt) {
      v16bf aq = load_a16x32(qh + mt * 16 * 256, 256, lane);
      v8f s[4];
#pragma unroll
      for (int ntv = 0; ntv < 4; ++ntv) {
        v8f z = {};
        s[ntv] = wmma_bf16(aq, bk[ntv], z);       // already scaled via Q
      }
      // row softmax over 64 cols: 4 tiles elementwise + 16-lane-half reductions
#pragma unroll
      for (int j = 0; j < 8; ++j) {
        const int row = mt * 16 + r0 + j;
        float e0 = s[0][j] + bh[row * 64 +  0 + nl];
        float e1 = s[1][j] + bh[row * 64 + 16 + nl];
        float e2 = s[2][j] + bh[row * 64 + 32 + nl];
        float e3 = s[3][j] + bh[row * 64 + 48 + nl];
        float mx = fmaxf(fmaxf(e0, e1), fmaxf(e2, e3));
#pragma unroll
        for (int msk = 1; msk < 16; msk <<= 1)
          mx = fmaxf(mx, __shfl_xor(mx, msk, 32));
        e0 = __expf(e0 - mx); e1 = __expf(e1 - mx);
        e2 = __expf(e2 - mx); e3 = __expf(e3 - mx);
        float sm = e0 + e1 + e2 + e3;
#pragma unroll
        for (int msk = 1; msk < 16; msk <<= 1)
          sm += __shfl_xor(sm, msk, 32);
        const float inv = 1.0f / sm;
        ph[row * 64 +  0 + nl] = (bf16_t)(e0 * inv);
        ph[row * 64 + 16 + nl] = (bf16_t)(e1 * inv);
        ph[row * 64 + 32 + nl] = (bf16_t)(e2 * inv);
        ph[row * 64 + 48 + nl] = (bf16_t)(e3 * inv);
      }
    }

    // O_h = P(64x64) @ V_h(64x32); V^T storage makes B-frags contiguous.
#pragma unroll
    for (int nt2 = 0; nt2 < 2; ++nt2) {
      v8f oacc[4] = {};
#pragma unroll
      for (int k2 = 0; k2 < 2; ++k2) {
        v16bf bv = load_b32x16(vt + (h * 32 + nt2 * 16) * 64 + k2 * 32, 64, lane);
        v16bf ap[4];
#pragma unroll
        for (int mt = 0; mt < 4; ++mt)
          ap[mt] = load_a16x32(ph + mt * 16 * 64 + k2 * 32, 64, lane);
#pragma unroll
        for (int mt = 0; mt < 4; ++mt)
          oacc[mt] = wmma_bf16(ap[mt], bv, oacc[mt]);
      }
#pragma unroll
      for (int mt = 0; mt < 4; ++mt)
#pragma unroll
        for (int j = 0; j < 8; ++j)
          os[(mt * 16 + r0 + j) * 256 + h * 32 + nt2 * 16 + nl] = (bf16_t)oacc[mt][j];
    }
  }
  __syncthreads();

  // ---- output projection: [64x256] x [256x256]; 16 N-tiles / 8 waves ----
  float* og = out + (size_t)blk * (60 * 256);
  for (int nt = wave; nt < 16; nt += 8) {
    v8f acc[4] = {};
    const bf16_t* wb = wo_t + nt * 16 * 256;
#pragma unroll
    for (int k = 0; k < 8; ++k) {
      v16bf bfr = load_b32x16(wb + k * 32, 256, lane);
      v16bf af[4];
#pragma unroll
      for (int mt = 0; mt < 4; ++mt)
        af[mt] = load_a16x32(os + mt * 16 * 256 + k * 32, 256, lane);
#pragma unroll
      for (int mt = 0; mt < 4; ++mt)
        acc[mt] = wmma_bf16(af[mt], bfr, acc[mt]);
    }
#pragma unroll
    for (int mt = 0; mt < 4; ++mt)
#pragma unroll
      for (int j = 0; j < 8; ++j) {
        const int t = mt * 16 + r0 + j;
        if (t < 60) og[t * 256 + nt * 16 + nl] = acc[mt][j];
      }
  }
}

// ---------- launcher ----------

extern "C" void kernel_launch(void* const* d_in, const int* in_sizes, int n_in,
                              void* d_out, int out_size, void* d_ws, size_t ws_size,
                              hipStream_t stream) {
  const float* x     = (const float*)d_in[0];   // [4096][60][256]
  const float* wqkv  = (const float*)d_in[1];   // [256][768]
  const float* wout  = (const float*)d_in[2];   // [256][256]
  const float* btab  = (const float*)d_in[3];   // [209][8]
  const int*   rpi   = (const int*)d_in[4];     // [60][60]
  float*       out   = (float*)d_out;           // [4096][60][256]

  // workspace layout
  bf16_t* wq_t      = (bf16_t*)d_ws;                               // 393216 B
  bf16_t* wo_t      = (bf16_t*)((char*)d_ws + 393216);             // 131072 B
  float*  bias_full = (float*)((char*)d_ws + 393216 + 131072);     // 131072 B

  prep_kernel<<<(768 * 256 + 255) / 256, 256, 0, stream>>>(
      wqkv, wout, btab, rpi, wq_t, wo_t, bias_full);

  const size_t lds_bytes = (size_t)(64 * 256 * 4 + 256 * 64 + 8 * 64 * 64) * sizeof(bf16_t);
  attn_win_kernel<<<4096, 256, lds_bytes, stream>>>(x, wq_t, wo_t, bias_full, out);
}
